// ClusteringLayer_53154515256373
// MI455X (gfx1250) — matrix operand order repaired and checked by reading, not compiled, not run on hardware
//
#include <hip/hip_runtime.h>

typedef __attribute__((ext_vector_type(16))) _Float16 v16h;
typedef __attribute__((ext_vector_type(4)))  _Float16 v4h;
typedef __attribute__((ext_vector_type(8)))  float    v8f;

#define AP   68      // padded LDS row stride in halves (64 + 4: bank-conflict-free, 8B aligned)
#define MBLK 64      // rows per workgroup
#define KC   256     // centroids
#define FD   64      // feature dim

// ---- fragment loaders (CDNA5 16-bit WMMA VGPR layouts, ISA 7.12.2) ----
// A 16x32 f16: lane L holds row M=L%16; half=L/16 selects K-subranges
//   element e -> K = kbase + 8*half + (e&7) + 16*(e>>3)
__device__ __forceinline__ v16h load_a_frag(const _Float16* Ah, int arow, int kbase, int half) {
  const v4h* p0 = (const v4h*)&Ah[arow * AP + kbase + 8 * half];       // K runs [kb+8h, +8)
  const v4h* p1 = (const v4h*)&Ah[arow * AP + kbase + 8 * half + 16];  // K runs [kb+8h+16, +8)
  v4h c0 = p0[0], c1 = p0[1], c2 = p1[0], c3 = p1[1];
  v16h a;
#pragma unroll
  for (int e = 0; e < 4; ++e) {
    a[e] = c0[e]; a[e + 4] = c1[e]; a[e + 8] = c2[e]; a[e + 12] = c3[e];
  }
  return a;
}

// B 32x16 f16: lane L holds column N=L%16; element e -> K = kbase + 8*half + (e&7) + 16*(e>>3)
__device__ __forceinline__ v16h load_b_frag(const _Float16* Ch, int col, int kbase, int half) {
  const v4h* p = (const v4h*)&Ch[col * AP + kbase + 8 * half];
  const v4h* q = (const v4h*)&Ch[col * AP + kbase + 8 * half + 16];
  v4h c0 = p[0], c1 = p[1], c2 = q[0], c3 = q[1];
  v16h b;
#pragma unroll
  for (int e = 0; e < 4; ++e) {
    b[e] = c0[e]; b[e + 4] = c1[e]; b[e + 8] = c2[e]; b[e + 12] = c3[e];
  }
  return b;
}

__global__ __launch_bounds__(256) void cluster_student_t_kernel(
    const float* __restrict__ X,   // (N, 64)
    const float* __restrict__ C,   // (256, 64)
    float* __restrict__ O)         // (N, 256)
{
  __shared__ _Float16 Ah[MBLK * AP];   // 8.7 KB
  __shared__ _Float16 Ch[KC * AP];     // 34.8 KB
  __shared__ float xsq[MBLK];
  __shared__ float csq[KC];
  __shared__ float rowsum[MBLK][2];    // one slot per column-half wave (deterministic, no atomics)

  const int    tid = threadIdx.x;
  const size_t R0  = (size_t)blockIdx.x * MBLK;

  // ---- stage X tile (64x64 fp32 -> f16 LDS), coalesced float4 loads ----
  {
    const float4* Xv = (const float4*)(X + R0 * FD);
#pragma unroll
    for (int p = 0; p < 4; ++p) {
      int f4 = tid + p * 256;                    // 0..1023
      float4 v = Xv[f4];
      int eb = f4 << 2, r = eb >> 6, k = eb & 63;
      *(v4h*)&Ah[r * AP + k] =
          (v4h){(_Float16)v.x, (_Float16)v.y, (_Float16)v.z, (_Float16)v.w};
    }
  }
  // ---- stage centroids (256x64 fp32 -> f16 LDS) ----
  {
    const float4* Cv = (const float4*)C;
#pragma unroll
    for (int p = 0; p < 16; ++p) {
      int f4 = tid + p * 256;                    // 0..4095
      float4 v = Cv[f4];
      int eb = f4 << 2, r = eb >> 6, k = eb & 63;
      *(v4h*)&Ch[r * AP + k] =
          (v4h){(_Float16)v.x, (_Float16)v.y, (_Float16)v.z, (_Float16)v.w};
    }
  }
  // ---- exact fp32 squared norms (L2-hot re-reads; keeps d2 accurate) ----
  if (tid < MBLK) {
    const float4* xr = (const float4*)(X + (R0 + tid) * FD);
    float s = 0.f;
#pragma unroll
    for (int i = 0; i < 16; ++i) { float4 v = xr[i]; s += v.x*v.x + v.y*v.y + v.z*v.z + v.w*v.w; }
    xsq[tid] = s;
  }
  {
    const float4* cr = (const float4*)(C + tid * FD);
    float s = 0.f;
#pragma unroll
    for (int i = 0; i < 16; ++i) { float4 v = cr[i]; s += v.x*v.x + v.y*v.y + v.z*v.z + v.w*v.w; }
    csq[tid] = s;
  }
  __syncthreads();

  // ---- per-wave tile assignment ----
  const int lane = tid & 31, wave = tid >> 5;
  const int half = lane >> 4, l16 = lane & 15;
  const int rowTile = wave >> 1;     // 0..3 -> rows [rowTile*16, +16)
  const int colHalf = wave & 1;      // 0..1 -> cols [colHalf*128, +128), 8 tiles of 16

  const v16h a0 = load_a_frag(Ah, rowTile * 16 + l16, 0,  half);
  const v16h a1 = load_a_frag(Ah, rowTile * 16 + l16, 32, half);

  int rl[8];
#pragma unroll
  for (int i = 0; i < 8; ++i) rl[i] = rowTile * 16 + i + 8 * half;   // C/D layout: M = i + 8*half

  float rsum[8];
#pragma unroll
  for (int i = 0; i < 8; ++i) rsum[i] = 0.f;

  v8f qt[8];

#pragma unroll
  for (int ct = 0; ct < 8; ++ct) {
    const int col0 = colHalf * 128 + ct * 16;
    const int ccol = col0 + l16;                 // B layout: N = lane & 15
    v16h b0 = load_b_frag(Ch, ccol, 0,  half);
    v16h b1 = load_b_frag(Ch, ccol, 32, half);

    v8f c = {};
    c = __builtin_amdgcn_wmma_f32_16x16x32_f16(false, a0, false, b0, (short)0, c, false, false);
    c = __builtin_amdgcn_wmma_f32_16x16x32_f16(false, a1, false, b1, (short)0, c, false, false);

    const float cs = csq[ccol];
#pragma unroll
    for (int i = 0; i < 8; ++i) {
      float d2 = xsq[rl[i]] + cs - 2.0f * c[i];
      d2 = fmaxf(d2, 0.0f);
      float q = 1.0f / (1.0f + d2);              // Student-t, ALPHA=1 -> exponent -1
      c[i] = q;
      // reduce q over the 16 columns held in this lane-half (masks < 16 never cross halves)
      float s = q;
      s += __shfl_xor(s, 1, 32);
      s += __shfl_xor(s, 2, 32);
      s += __shfl_xor(s, 4, 32);
      s += __shfl_xor(s, 8, 32);
      rsum[i] += s;
    }
    qt[ct] = c;
  }

  if (l16 == 0) {                                // lanes 0 and 16: unique (row, colHalf) slots
#pragma unroll
    for (int i = 0; i < 8; ++i) rowsum[rl[i]][colHalf] = rsum[i];
  }
  __syncthreads();

  float inv[8];
#pragma unroll
  for (int i = 0; i < 8; ++i) inv[i] = 1.0f / (rowsum[rl[i]][0] + rowsum[rl[i]][1]);

#pragma unroll
  for (int ct = 0; ct < 8; ++ct) {
    const int col0 = colHalf * 128 + ct * 16;
#pragma unroll
    for (int i = 0; i < 8; ++i) {
      O[(R0 + rl[i]) * (size_t)KC + col0 + l16] = qt[ct][i] * inv[i];
    }
  }
}

extern "C" void kernel_launch(void* const* d_in, const int* in_sizes, int n_in,
                              void* d_out, int out_size, void* d_ws, size_t ws_size,
                              hipStream_t stream) {
  const float* X = (const float*)d_in[0];   // (524288, 64) fp32
  const float* C = (const float*)d_in[1];   // (256, 64) fp32
  float* O = (float*)d_out;                 // (524288, 256) fp32

  const int N = in_sizes[0] / FD;           // 524288
  const int nblocks = N / MBLK;             // 8192
  cluster_student_t_kernel<<<nblocks, 256, 0, stream>>>(X, C, O);
}
